// CrossGate2_17497696764338
// MI455X (gfx1250) — matrix operand order repaired and checked by reading, not compiled, not run on hardware
//
#include <hip/hip_runtime.h>
#include <math.h>

// ---- CDNA5 WMMA vector types ----
typedef __attribute__((ext_vector_type(16))) __bf16 v16bf;
typedef __attribute__((ext_vector_type(8)))  __bf16 v8bf;
typedef __attribute__((ext_vector_type(8)))  float  v8f;

#define C_DIM 256
#define B_DIM 4
#define T_DIM 4
#define AH 32
#define AW 32
#define VH 56
#define VW 56
#define AHW (AH*AW)        // 1024
#define VHW (VH*VW)        // 3136
#define VB  (B_DIM*T_DIM)  // 16
#define EPSV 1e-5f

// float -> bf16, round-to-nearest-even (stored as ushort; reinterpret as __bf16 for WMMA)
__device__ __forceinline__ unsigned short f2bf(float f) {
    unsigned int u = __float_as_uint(f);
    unsigned int r = (u + 0x7fffu + ((u >> 16) & 1u)) >> 16;
    return (unsigned short)r;
}

// ---------------- GroupNorm(1,C) stats: per-sample mean / rstd ----------------
__global__ void gn_stats_kernel(const float* __restrict__ x, float* __restrict__ stats,
                                int per_sample) {
    __shared__ float s1[256], s2[256];
    int s = blockIdx.x;
    const float* xs = x + (size_t)s * per_sample;
    float a1 = 0.f, a2 = 0.f;
    for (int i = threadIdx.x; i < per_sample; i += blockDim.x) {
        float v = xs[i]; a1 += v; a2 += v * v;
    }
    s1[threadIdx.x] = a1; s2[threadIdx.x] = a2;
    __syncthreads();
    for (int off = 128; off > 0; off >>= 1) {
        if ((int)threadIdx.x < off) {
            s1[threadIdx.x] += s1[threadIdx.x + off];
            s2[threadIdx.x] += s2[threadIdx.x + off];
        }
        __syncthreads();
    }
    if (threadIdx.x == 0) {
        float mu  = s1[0] / (float)per_sample;
        float var = s2[0] / (float)per_sample - mu * mu;
        stats[s * 2 + 0] = mu;
        stats[s * 2 + 1] = rsqrtf(var + EPSV);
    }
}

// ---- fused GN affine + (dw7 pad3 dil1 + dw3 pad3 dil3) as one 7x7 stencil ----
// one block per (channel, image) plane; plane staged in LDS; out = bf16 NHWC
__global__ void gn_dw_kernel(const float* __restrict__ x, const float* __restrict__ stats,
                             const float* __restrict__ gn_w, const float* __restrict__ gn_b,
                             const float* __restrict__ dw7, const float* __restrict__ dw3,
                             unsigned short* __restrict__ preh, int H, int W) {
    extern __shared__ float sm[];
    int c = blockIdx.x, img = blockIdx.y;
    int HWp = H * W;
    float mu = stats[img * 2], rstd = stats[img * 2 + 1];
    float gw = gn_w[c] * rstd;
    float gb = gn_b[c] - mu * gw;
    const float* xp = x + ((size_t)img * C_DIM + c) * HWp;
    for (int i = threadIdx.x; i < HWp; i += blockDim.x) sm[i] = xp[i] * gw + gb;

    float Kc[49];
#pragma unroll
    for (int ky = 0; ky < 7; ky++)
#pragma unroll
        for (int kx = 0; kx < 7; kx++) {
            float k = dw7[c * 49 + ky * 7 + kx];
            if ((ky % 3 == 0) && (kx % 3 == 0)) k += dw3[c * 9 + (ky / 3) * 3 + (kx / 3)];
            Kc[ky * 7 + kx] = k;
        }
    __syncthreads();

    unsigned short* op = preh + (size_t)img * HWp * C_DIM + c;
    for (int p = threadIdx.x; p < HWp; p += blockDim.x) {
        int h = p / W, w = p % W;
        float s = 0.f;
#pragma unroll
        for (int ky = 0; ky < 7; ky++) {
            int ih = h + ky - 3;
            if ((unsigned)ih < (unsigned)H) {
#pragma unroll
                for (int kx = 0; kx < 7; kx++) {
                    int iw = w + kx - 3;
                    if ((unsigned)iw < (unsigned)W) s += sm[ih * W + iw] * Kc[ky * 7 + kx];
                }
            }
        }
        op[(size_t)p * C_DIM] = f2bf(s);
    }
}

// ---------------- f32 -> bf16 weight conversion ----------------
__global__ void f32_to_bf16_kernel(const float* __restrict__ src,
                                   unsigned short* __restrict__ dst, int n) {
    int i = blockIdx.x * blockDim.x + threadIdx.x;
    if (i < n) dst[i] = f2bf(src[i]);
}

// ---- T-mean + adaptive avg pool 56x56 -> 32x32, out bf16 NHWC ----
__global__ void pool_mean_kernel(const float* __restrict__ v_feat,
                                 unsigned short* __restrict__ v_for_a) {
    extern __shared__ float sm[];  // VHW floats
    int c = blockIdx.x, b = blockIdx.y;
    for (int p = threadIdx.x; p < VHW; p += blockDim.x) {
        float s = 0.f;
#pragma unroll
        for (int t = 0; t < T_DIM; t++)
            s += v_feat[(((size_t)(b * T_DIM + t)) * C_DIM + c) * VHW + p];
        sm[p] = s * (1.f / T_DIM);
    }
    __syncthreads();
    for (int op = threadIdx.x; op < AHW; op += blockDim.x) {
        int oh = op / AW, ow = op % AW;
        int sh = (oh * VH) / AH, eh = ((oh + 1) * VH + AH - 1) / AH;
        int sw = (ow * VW) / AW, ew = ((ow + 1) * VW + AW - 1) / AW;
        float acc = 0.f;
        for (int ih = sh; ih < eh; ih++)
            for (int iw = sw; iw < ew; iw++) acc += sm[ih * VW + iw];
        acc /= (float)((eh - sh) * (ew - sw));
        v_for_a[((size_t)b * AHW + op) * C_DIM + c] = f2bf(acc);
    }
}

// ---- bilinear resize 32x32 -> 56x56 (align_corners=False), out bf16 NHWC ----
__global__ void bilinear_kernel(const float* __restrict__ a_feat,
                                unsigned short* __restrict__ a_for_v) {
    extern __shared__ float sm[];  // AHW floats
    int c = blockIdx.x, b = blockIdx.y;
    const float* ap = a_feat + ((size_t)b * C_DIM + c) * AHW;
    for (int i = threadIdx.x; i < AHW; i += blockDim.x) sm[i] = ap[i];
    __syncthreads();
    const float sc_h = (float)AH / (float)VH, sc_w = (float)AW / (float)VW;
    for (int op = threadIdx.x; op < VHW; op += blockDim.x) {
        int oh = op / VW, ow = op % VW;
        float shf = fminf(fmaxf((oh + 0.5f) * sc_h - 0.5f, 0.f), (float)(AH - 1));
        float swf = fminf(fmaxf((ow + 0.5f) * sc_w - 0.5f, 0.f), (float)(AW - 1));
        int h0 = (int)floorf(shf); int h1 = min(h0 + 1, AH - 1); float wh = shf - (float)h0;
        int w0 = (int)floorf(swf); int w1 = min(w0 + 1, AW - 1); float ww = swf - (float)w0;
        float val = (1.f - wh) * ((1.f - ww) * sm[h0 * AW + w0] + ww * sm[h0 * AW + w1])
                  +        wh  * ((1.f - ww) * sm[h1 * AW + w0] + ww * sm[h1 * AW + w1]);
        a_for_v[((size_t)b * VHW + op) * C_DIM + c] = f2bf(val);
    }
}

// ---------------- WMMA bf16 GEMM: out[o,n] = act( sum_c W[o,c]*X[n,c] + bias[o] )
// block = 128 threads (4 wave32); wave w covers rows [64w,64w+64); block covers 32 cols.
// Wave tile 64x32 = 4x2 v8f accumulators (64 VGPRs) -> no scratch spills.
// MODE: 0/1 -> gelu, store f32 NCHW feat; 2 -> sigmoid gate, write a_out;
//       3 -> sigmoid gate, broadcast over T, write v_out.
template <int MODE>
__global__ __launch_bounds__(128, 1) void gemm_wmma_kernel(
    const unsigned short* __restrict__ Xh,    // [img][N][256] bf16
    const unsigned short* __restrict__ Wh,    // [256][256]    bf16
    const float* __restrict__ bias,           // [256]
    int N,
    float* __restrict__ out_feat,             // MODE 0/1
    const float* __restrict__ resid,          // MODE 2/3
    const float* __restrict__ feat,           // MODE 2/3
    const float* __restrict__ scale_p,        // MODE 2/3
    float* __restrict__ out_final)            // MODE 2/3
{
    const int lane   = (int)(threadIdx.x & 31);
    const int wave   = (int)(threadIdx.x >> 5);
    const int img    = (int)blockIdx.y;
    const int n_base = (int)blockIdx.x * 32;
    const int m_base = wave * 64;
    const int lmod   = lane & 15;
    const int lhalf  = lane >> 4;

    const unsigned short* Ximg = Xh + (size_t)img * N * C_DIM;

    v8f acc[4][2];
#pragma unroll
    for (int mi = 0; mi < 4; mi++)
#pragma unroll
        for (int ni = 0; ni < 2; ni++)
            acc[mi][ni] = (v8f){0.f, 0.f, 0.f, 0.f, 0.f, 0.f, 0.f, 0.f};

    const unsigned short* wbase = Wh + (size_t)(m_base + lmod) * C_DIM + 16 * lhalf;
    const unsigned short* xbase = Ximg + (size_t)(n_base + lmod) * C_DIM + 16 * lhalf;

    for (int k0 = 0; k0 < C_DIM; k0 += 32) {
        v16bf aF[4], bF[2];
        // A operand, contiguous k order (same as B): lane-half selects K half; elems 0..15 -> K = k0+16*lhalf+j
#pragma unroll
        for (int mi = 0; mi < 4; mi++) {
            const unsigned short* wrow = wbase + (size_t)(mi * 16) * C_DIM + k0;
            v8bf alo = *(const v8bf*)(wrow);
            v8bf ahi = *(const v8bf*)(wrow + 8);
            aF[mi] = __builtin_shufflevector(alo, ahi, 0, 1, 2, 3, 4, 5, 6, 7,
                                             8, 9, 10, 11, 12, 13, 14, 15);
        }
        // B operand, contiguous k order (same as A): lanes 0-15 hold K=k0..k0+15, lanes 16-31 K=k0+16..k0+31
#pragma unroll
        for (int ni = 0; ni < 2; ni++) {
            bF[ni] = *(const v16bf*)(xbase + (size_t)(ni * 16) * C_DIM + k0);
        }
#pragma unroll
        for (int mi = 0; mi < 4; mi++)
#pragma unroll
            for (int ni = 0; ni < 2; ni++)
                acc[mi][ni] = __builtin_amdgcn_wmma_f32_16x16x32_bf16(
                    false, aF[mi], false, bF[ni], (short)0, acc[mi][ni], false, false);
    }

    const float sc = (MODE >= 2) ? scale_p[0] : 0.f;
#pragma unroll
    for (int mi = 0; mi < 4; mi++) {
#pragma unroll
        for (int ni = 0; ni < 2; ni++) {
            int n = n_base + ni * 16 + lmod;
#pragma unroll
            for (int r = 0; r < 8; r++) {
                int m = m_base + mi * 16 + r + 8 * lhalf;  // D: VGPR r -> M=r (+8 upper half)
                float val = acc[mi][ni][r] + bias[m];
                if (MODE <= 1) {
                    float g = 0.5f * val * (1.f + erff(val * 0.70710678118654752f));
                    out_feat[((size_t)img * C_DIM + m) * N + n] = g;
                } else if (MODE == 2) {
                    float g = 1.f / (1.f + expf(-val));
                    size_t idx = ((size_t)img * C_DIM + m) * N + n;
                    out_final[idx] = resid[idx] + sc * feat[idx] * g;
                } else {
                    float g = 1.f / (1.f + expf(-val));
#pragma unroll
                    for (int t = 0; t < T_DIM; t++) {
                        size_t idx = (((size_t)(img * T_DIM + t)) * C_DIM + m) * N + n;
                        out_final[idx] = resid[idx] + sc * feat[idx] * g;
                    }
                }
            }
        }
    }
}

extern "C" void kernel_launch(void* const* d_in, const int* in_sizes, int n_in,
                              void* d_out, int out_size, void* d_ws, size_t ws_size,
                              hipStream_t stream) {
    (void)in_sizes; (void)n_in; (void)out_size; (void)ws_size;
    const float* a        = (const float*)d_in[0];
    const float* v        = (const float*)d_in[1];
    const float* a_gn_w   = (const float*)d_in[2];
    const float* a_gn_b   = (const float*)d_in[3];
    const float* a_dw7    = (const float*)d_in[4];
    const float* a_dw3    = (const float*)d_in[5];
    const float* a_pw_w   = (const float*)d_in[6];
    const float* a_pw_b   = (const float*)d_in[7];
    const float* v_gn_w   = (const float*)d_in[8];
    const float* v_gn_b   = (const float*)d_in[9];
    const float* v_dw7    = (const float*)d_in[10];
    const float* v_dw3    = (const float*)d_in[11];
    const float* v_pw_w   = (const float*)d_in[12];
    const float* v_pw_b   = (const float*)d_in[13];
    const float* gate_a_w = (const float*)d_in[14];
    const float* gate_a_b = (const float*)d_in[15];
    const float* gate_v_w = (const float*)d_in[16];
    const float* gate_v_b = (const float*)d_in[17];
    const float* scale_a  = (const float*)d_in[18];
    const float* scale_v  = (const float*)d_in[19];

    // ---- workspace layout (256B aligned segments) ----
    char* ws = (char*)d_ws;
    size_t off = 0;
    auto take = [&](size_t bytes) -> char* {
        char* p = ws + off;
        off = (off + bytes + 255) & ~((size_t)255);
        return p;
    };
    float*          a_stats  = (float*)take(B_DIM * 2 * sizeof(float));
    float*          v_stats  = (float*)take(VB * 2 * sizeof(float));
    unsigned short* wh_pw_a  = (unsigned short*)take((size_t)C_DIM * C_DIM * 2);
    unsigned short* wh_pw_v  = (unsigned short*)take((size_t)C_DIM * C_DIM * 2);
    unsigned short* wh_ga    = (unsigned short*)take((size_t)C_DIM * C_DIM * 2);
    unsigned short* wh_gv    = (unsigned short*)take((size_t)C_DIM * C_DIM * 2);
    unsigned short* a_preh   = (unsigned short*)take((size_t)B_DIM * AHW * C_DIM * 2);
    unsigned short* v_preh   = (unsigned short*)take((size_t)VB * VHW * C_DIM * 2);
    float*          a_feat   = (float*)take((size_t)B_DIM * C_DIM * AHW * 4);
    float*          v_feat   = (float*)take((size_t)VB * C_DIM * VHW * 4);
    unsigned short* v_for_a  = (unsigned short*)take((size_t)B_DIM * AHW * C_DIM * 2);
    unsigned short* a_for_v  = (unsigned short*)take((size_t)B_DIM * VHW * C_DIM * 2);

    float* a_out = (float*)d_out;
    float* v_out = (float*)d_out + (size_t)B_DIM * C_DIM * AHW;

    const int wN = C_DIM * C_DIM;  // 65536
    f32_to_bf16_kernel<<<wN / 256, 256, 0, stream>>>(a_pw_w,   wh_pw_a, wN);
    f32_to_bf16_kernel<<<wN / 256, 256, 0, stream>>>(v_pw_w,   wh_pw_v, wN);
    f32_to_bf16_kernel<<<wN / 256, 256, 0, stream>>>(gate_a_w, wh_ga,   wN);
    f32_to_bf16_kernel<<<wN / 256, 256, 0, stream>>>(gate_v_w, wh_gv,   wN);

    gn_stats_kernel<<<B_DIM, 256, 0, stream>>>(a, a_stats, C_DIM * AHW);
    gn_stats_kernel<<<VB,    256, 0, stream>>>(v, v_stats, C_DIM * VHW);

    gn_dw_kernel<<<dim3(C_DIM, B_DIM), 256, AHW * sizeof(float), stream>>>(
        a, a_stats, a_gn_w, a_gn_b, a_dw7, a_dw3, a_preh, AH, AW);
    gn_dw_kernel<<<dim3(C_DIM, VB), 256, VHW * sizeof(float), stream>>>(
        v, v_stats, v_gn_w, v_gn_b, v_dw7, v_dw3, v_preh, VH, VW);

    // a_feat = gelu(pwconv(a_pre))    : N=1024, 4 images
    gemm_wmma_kernel<0><<<dim3(AHW / 32, B_DIM), 128, 0, stream>>>(
        a_preh, wh_pw_a, a_pw_b, AHW, a_feat, nullptr, nullptr, nullptr, nullptr);
    // v_feat = gelu(pwconv(v_pre))    : N=3136, 16 images
    gemm_wmma_kernel<1><<<dim3(VHW / 32, VB), 128, 0, stream>>>(
        v_preh, wh_pw_v, v_pw_b, VHW, v_feat, nullptr, nullptr, nullptr, nullptr);

    pool_mean_kernel<<<dim3(C_DIM, B_DIM), 256, VHW * sizeof(float), stream>>>(v_feat, v_for_a);
    bilinear_kernel<<<dim3(C_DIM, B_DIM), 256, AHW * sizeof(float), stream>>>(a_feat, a_for_v);

    // a_out = a + scale_a * a_feat * sigmoid(pwconv(v_for_a))
    gemm_wmma_kernel<2><<<dim3(AHW / 32, B_DIM), 128, 0, stream>>>(
        v_for_a, wh_ga, gate_a_b, AHW, nullptr, a, a_feat, scale_a, a_out);
    // v_out = v + scale_v * v_feat * sigmoid(pwconv(a_for_v))  (gate broadcast over T)
    gemm_wmma_kernel<3><<<dim3(VHW / 32, B_DIM), 128, 0, stream>>>(
        a_for_v, wh_gv, gate_v_b, VHW, nullptr, v, v_feat, scale_v, v_out);
}
